// LSTMEncoder_65970697666968
// MI455X (gfx1250) — matrix operand order repaired and checked by reading, not compiled, not run on hardware
//
#include <hip/hip_runtime.h>
#include <hip/hip_bf16.h>

// ---------------------------------------------------------------------------
// LSTM encoder for MI455X (gfx1250), wave32 + WMMA bf16.
//
// Persistent-kernel design: the serial T=512 recurrence is the bottleneck
// (weights are L2/LDS resident; HBM traffic is trivial at 23.3 TB/s), so we
// run one persistent grid with a grid barrier per time step, weights in LDS,
// cell state c in WMMA-accumulator registers, and h ping-ponged through two
// small bf16 activation buffers in the workspace.
//
// Round-3 refinement: LDS weights re-laid out hidden-unit-major so the four
// gate rows of a unit are adjacent -> every ds_load_b128 in the k-loop uses a
// 16-bit immediate offset from ONE per-lane base register (round 2 still had
// per-iteration v_add chains because gates 2/3 sat beyond the 64 KB DS offset
// range). Group stride 12368 B = 3092 dwords (3092%64==20) keeps the 16-lane
// halves bank-conflict-free for b128 accesses.
// ---------------------------------------------------------------------------

typedef __attribute__((ext_vector_type(16))) __bf16 v16bf;
typedef __attribute__((ext_vector_type(8)))  __bf16 v8bf;
typedef __attribute__((ext_vector_type(4)))  __bf16 v4bf;
typedef __attribute__((ext_vector_type(4)))  float  v4f;
typedef __attribute__((ext_vector_type(8)))  float  v8f;

#define B_SZ   128
#define T_SZ   512
#define D_SZ   512
#define H_SZ   1024
#define G4     4096          // 4*H
#define KC     1536          // D + H (fused [x_t | h] GEMM K)
#define LDA    1536          // activation buffer leading dim (bf16 elems)
#define LDSROW 1544          // one weight row in LDS (bf16 elems)
#define GSZ    6184          // hidden-unit group stride = 4*LDSROW + 8 elems
#define NWG    64
#define NTHR   256
#define HID_PER_WG 16        // H / NWG
#define L2OUT  256

// ---- ISA-exact fragment loaders (CDNA5 ISA §7.12.2) -------------
// laneBase already includes the per-lane offset; elemOff is uniform.
// A 16x32 bf16: lane l -> M = l&15; elems 0..7 <-> K = 8*(l>>4)+e,
//                              elems 8..15 <-> K = 16 + 8*(l>>4)+(e-8).
// laneBase_A = row(m)*ld + 8*(l>>4); second half at +16 elements.
__device__ __forceinline__ v16bf load_a_frag(const __bf16* laneBase, int elemOff) {
    v8bf lo = *(const v8bf*)(laneBase + elemOff);
    v8bf hb = *(const v8bf*)(laneBase + elemOff + 16);
    return __builtin_shufflevector(lo, hb, 0,1,2,3,4,5,6,7,8,9,10,11,12,13,14,15);
}

// B 32x16 bf16: lane l -> N = l&15; same K order as A (8*(l>>4)+e, 16+8*(l>>4)+(e-8)).
// laneBase_B = row(n) + 8*(l>>4); second half at +16 elements.
__device__ __forceinline__ v16bf load_b_frag(const __bf16* laneBase, int elemOff) {
    v8bf b0 = *(const v8bf*)(laneBase + elemOff);
    v8bf b1 = *(const v8bf*)(laneBase + elemOff + 16);
    return __builtin_shufflevector(b0, b1, 0,1,2,3,4,5,6,7,8,9,10,11,12,13,14,15);
}

__device__ __forceinline__ v8f wmma_bf16(v16bf a, v16bf b, v8f c) {
    return __builtin_amdgcn_wmma_f32_16x16x32_bf16(false, a, false, b,
                                                   (short)0, c, false, false);
}

__device__ __forceinline__ float sigm_f(float x) {
    float e = __expf(-fabsf(x));
    float s = 1.0f / (1.0f + e);
    return x >= 0.0f ? s : 1.0f - s;
}
__device__ __forceinline__ float tanh_f(float x) {
    float e = __expf(-2.0f * fabsf(x));
    float t = (1.0f - e) / (1.0f + e);
    return x >= 0.0f ? t : -t;
}

#define V8F_ZERO {0.f,0.f,0.f,0.f,0.f,0.f,0.f,0.f}

// ---------------------------------------------------------------------------
// Kernel 1: prep — bias sum, bf16 W_out, initial activation buffer
// abuf0 = [ bf16(x[:, T-1, :]) | zeros(h0) ],  row-major [B][KC].
// ---------------------------------------------------------------------------
__global__ void __launch_bounds__(256)
prep_kernel(const float* __restrict__ x, const float* __restrict__ b_ih,
            const float* __restrict__ b_hh, const float* __restrict__ W_out,
            float* __restrict__ bsum, __bf16* __restrict__ WoutBf,
            __bf16* __restrict__ abuf0)
{
    const int gtid = blockIdx.x * blockDim.x + threadIdx.x;
    const int nth  = gridDim.x * blockDim.x;
    for (int i = gtid; i < G4; i += nth) bsum[i] = b_ih[i] + b_hh[i];
    for (int i = gtid; i < L2OUT * H_SZ; i += nth) WoutBf[i] = (__bf16)W_out[i];
    for (int i = gtid; i < B_SZ * KC; i += nth) {
        int b = i / KC, c = i - b * KC;
        float v = (c < D_SZ) ? x[(b * T_SZ + (T_SZ - 1)) * D_SZ + c] : 0.0f;
        abuf0[i] = (__bf16)v;
    }
}

// ---------------------------------------------------------------------------
// Kernel 2: persistent LSTM recurrence.
// Each of NWG=64 workgroups owns 16 hidden units -> 64 weight rows
// (i,f,g,o x 16) x K=1536, staged once into ~193 KB of LDS (bf16),
// hidden-unit-major: lds[rr*GSZ + g*LDSROW + k].
// Wave w computes the [16 batch x 64 gate-col] tile; c state lives in a v8f.
// One grid barrier (atomic counter + s_sleep spin) per time step.
// NOTE: assumes all 64 WGs are co-resident (1 WG / WGP at 193 KB LDS).
// ---------------------------------------------------------------------------
__global__ void __launch_bounds__(NTHR)
lstm_persistent(const float* __restrict__ x, const float* __restrict__ W_ih,
                const float* __restrict__ W_hh, const float* __restrict__ bsum,
                __bf16* __restrict__ abuf0, __bf16* __restrict__ abuf1,
                unsigned* __restrict__ counter)
{
    extern __shared__ __bf16 ldsW[];           // [16 units][4 gates][LDSROW]
    const int tid  = threadIdx.x;
    const int wg   = blockIdx.x;
    const int hid0 = wg * HID_PER_WG;

    // ---- stage combined [W_ih | W_hh] rows for this WG into LDS (once) ----
    // idx -> (rr, g, k); consecutive idx = consecutive k (coalesced reads).
    for (int idx = tid; idx < 64 * KC; idx += NTHR) {
        int rr  = idx / (4 * KC);
        int rem = idx - rr * (4 * KC);
        int g   = rem / KC;
        int k   = rem - g * KC;
        int grow = g * H_SZ + hid0 + rr;       // global gate row (i,f,g,o order)
        float w = (k < D_SZ) ? W_ih[grow * D_SZ + k]
                             : W_hh[grow * H_SZ + (k - D_SZ)];
        ldsW[rr * GSZ + g * LDSROW + k] = (__bf16)w;
    }
    __syncthreads();

    const int wave = tid >> 5;
    const int lane = tid & 31;
    const int hi   = lane >> 4;
    const int n    = lane & 15;
    const int gtid = wg * NTHR + tid;

    // per-lane gate biases for hidden unit hid0+n (column of our D tile)
    const float bi = bsum[0 * H_SZ + hid0 + n];
    const float bf = bsum[1 * H_SZ + hid0 + n];
    const float bg = bsum[2 * H_SZ + hid0 + n];
    const float bo = bsum[3 * H_SZ + hid0 + n];

    // ---- loop-invariant per-lane base pointers (hoisted address math) -----
    // Single B base; gate g at immediate offset g*LDSROW (max ~12.4 KB).
    const __bf16* wbase = ldsW + n * GSZ + hi * 8;
    // A fragments: our 16 batch rows, lane row m=n, lane offset 8*hi; one base
    // per ping-pong buffer.
    const __bf16* aBase0 = abuf0 + (wave * 16 + n) * LDA + hi * 8;
    const __bf16* aBase1 = abuf1 + (wave * 16 + n) * LDA + hi * 8;
    // h stores: D-layout row (wave*16 + 8*hi + v), col D_SZ + hid0 + n; the
    // v-dependent part (v*LDA*2B = v*3072B) folds into the store offset imm.
    __bf16* hBase0 = abuf0 + (wave * 16 + 8 * hi) * LDA + D_SZ + hid0 + n;
    __bf16* hBase1 = abuf1 + (wave * 16 + 8 * hi) * LDA + D_SZ + hid0 + n;
    // x staging: one float4 group per thread per step (B*D/4 = 16384 = grid).
    const int xb = gtid >> 7;                 // batch row (128 groups per row)
    const int xd = (gtid & 127) * 4;          // d offset
    const float* xsrc = x + (xb * T_SZ) * D_SZ + xd;
    __bf16* xdst0 = abuf0 + xb * LDA + xd;
    __bf16* xdst1 = abuf1 + xb * LDA + xd;

    v8f creg = V8F_ZERO;                       // c state, C/D-fragment layout

    for (int t = 0; t < T_SZ; ++t) {
        const bool odd = (t & 1) != 0;
        const __bf16* aBase = odd ? aBase1 : aBase0;  // read [x_t | h_{t-1}]
        __bf16* hBase = odd ? hBase0 : hBase1;        // write h_t (other buf)
        __bf16* xdst  = odd ? xdst0  : xdst1;         // write x_{t+1}

        v8f acc0 = V8F_ZERO, acc1 = V8F_ZERO, acc2 = V8F_ZERO, acc3 = V8F_ZERO;

        #pragma unroll 4
        for (int kk = 0; kk < KC / 32; ++kk) {
            const int eo = kk * 32;
            v16bf a  = load_a_frag(aBase, eo);
            v16bf w0 = load_b_frag(wbase, 0 * LDSROW + eo);
            acc0 = wmma_bf16(a, w0, acc0);
            v16bf w1 = load_b_frag(wbase, 1 * LDSROW + eo);
            acc1 = wmma_bf16(a, w1, acc1);
            v16bf w2 = load_b_frag(wbase, 2 * LDSROW + eo);
            acc2 = wmma_bf16(a, w2, acc2);
            v16bf w3 = load_b_frag(wbase, 3 * LDSROW + eo);
            acc3 = wmma_bf16(a, w3, acc3);
        }

        // ---- gates -> c,h (all in D-fragment layout), write h_t as bf16 ----
        #pragma unroll
        for (int v = 0; v < 8; ++v) {
            float iv = sigm_f(acc0[v] + bi);
            float fv = sigm_f(acc1[v] + bf);
            float gv = tanh_f(acc2[v] + bg);
            float ov = sigm_f(acc3[v] + bo);
            float cv = fv * creg[v] + iv * gv;
            creg[v] = cv;
            float hv = ov * tanh_f(cv);
            hBase[v * LDA] = (__bf16)hv;              // offset imm: v*3072 B
        }

        // ---- stage x_{t+1} (reversed time): one f32x4 -> bf16x4 per thread -
        if (t < T_SZ - 1) {
            const int tt = T_SZ - 2 - t;              // = T-1-(t+1)
            v4f xv = *(const v4f*)(xsrc + tt * D_SZ);
            v4bf xp = { (__bf16)xv.x, (__bf16)xv.y, (__bf16)xv.z, (__bf16)xv.w };
            *(v4bf*)xdst = xp;
        }

        // ---- grid barrier: monotone counter, target = NWG*(t+1) -----------
        __syncthreads();
        if (tid == 0) {
            __threadfence();                          // release our writes
            atomicAdd(counter, 1u);
            const unsigned target = (unsigned)gridDim.x * (unsigned)(t + 1);
            volatile unsigned* vc = counter;
            while (*vc < target) { __builtin_amdgcn_s_sleep(1); }
        }
        __syncthreads();
        __threadfence();                              // acquire peer writes
    }
}

// ---------------------------------------------------------------------------
// Kernel 3: head — out = h_T @ W_out^T + b_out, one WMMA tile per wave.
// 128x256 output = 8x16 tiles = 128 waves = 16 blocks x 8 waves.
// ---------------------------------------------------------------------------
__global__ void __launch_bounds__(256)
head_kernel(const __bf16* __restrict__ abuf0, const __bf16* __restrict__ WoutBf,
            const float* __restrict__ b_out, float* __restrict__ out)
{
    const int wave = threadIdx.x >> 5, lane = threadIdx.x & 31;
    const int gw = blockIdx.x * 8 + wave;   // 0..127
    const int mt = gw >> 4, nt = gw & 15;
    const int b0 = mt * 16, j0 = nt * 16;
    const int hi = lane >> 4, n = lane & 15;

    v8f acc = V8F_ZERO;
    const __bf16* aBase = abuf0 + (b0 + n) * LDA + D_SZ + hi * 8;
    const __bf16* bBase = WoutBf + (j0 + n) * H_SZ + hi * 8;

    #pragma unroll 4
    for (int kk = 0; kk < H_SZ / 32; ++kk) {
        v16bf a = load_a_frag(aBase, kk * 32);
        v16bf w = load_b_frag(bBase, kk * 32);
        acc = wmma_bf16(a, w, acc);
    }

    const float bias = b_out[j0 + n];
    #pragma unroll
    for (int v = 0; v < 8; ++v)
        out[(b0 + v + 8 * hi) * L2OUT + j0 + n] = acc[v] + bias;
}

// ---------------------------------------------------------------------------
// Workspace layout (bytes):
//   [0,512)        barrier counter (+pad)       — reset each launch
//   [512,16896)    bsum f32 [4096]
//   [17408,541696) W_out bf16 [256][1024]
//   [541696,...)   abuf0 bf16 [128][1536]  (384 KB)
//   [934912,...)   abuf1 bf16 [128][1536]  (384 KB)
// Total ~1.27 MB.
// ---------------------------------------------------------------------------
extern "C" void kernel_launch(void* const* d_in, const int* in_sizes, int n_in,
                              void* d_out, int out_size, void* d_ws, size_t ws_size,
                              hipStream_t stream)
{
    const float* x     = (const float*)d_in[0];
    const float* W_ih  = (const float*)d_in[1];
    const float* W_hh  = (const float*)d_in[2];
    const float* b_ih  = (const float*)d_in[3];
    const float* b_hh  = (const float*)d_in[4];
    const float* W_out = (const float*)d_in[5];
    const float* b_out = (const float*)d_in[6];
    float* out = (float*)d_out;

    char* ws = (char*)d_ws;
    unsigned* counter = (unsigned*)(ws);
    float*    bsum    = (float*)  (ws + 512);
    __bf16*   WoutBf  = (__bf16*) (ws + 17408);
    __bf16*   abuf0   = (__bf16*) (ws + 541696);
    __bf16*   abuf1   = (__bf16*) (ws + 934912);

    // reset grid-barrier counter (graph-capturable)
    hipMemsetAsync(ws, 0, 512, stream);

    prep_kernel<<<256, 256, 0, stream>>>(x, b_ih, b_hh, W_out, bsum, WoutBf, abuf0);

    const size_t ldsBytes = (size_t)16 * GSZ * sizeof(__bf16);  // 197888 B
    lstm_persistent<<<NWG, NTHR, ldsBytes, stream>>>(x, W_ih, W_hh, bsum,
                                                     abuf0, abuf1, counter);

    head_kernel<<<16, 256, 0, stream>>>(abuf0, WoutBf, b_out, out);
}
